// RecurrentNatureCNN_52226802319465
// MI455X (gfx1250) — matrix operand order repaired and checked by reading, not compiled, not run on hardware
//
#include <hip/hip_runtime.h>

typedef __attribute__((ext_vector_type(16))) _Float16 v16h;
typedef __attribute__((ext_vector_type(8)))  _Float16 v8h;
typedef __attribute__((ext_vector_type(4)))  _Float16 v4h;
typedef __attribute__((ext_vector_type(8)))  float    v8f;

#define DEVINL __device__ __forceinline__

// ---------------------------------------------------------------------------
// CDNA5 async copy: global -> LDS without touching VGPRs (ASYNCcnt tracked).
// ---------------------------------------------------------------------------
DEVINL void async_load_b128(unsigned lds_off, unsigned gl_off, const void* base) {
  asm volatile("global_load_async_to_lds_b128 %0, %1, %2"
               :: "v"(lds_off), "v"(gl_off),
                  "s"((unsigned long long)(size_t)base) : "memory");
}
DEVINL void async_load_b64(unsigned lds_off, unsigned gl_off, const void* base) {
  asm volatile("global_load_async_to_lds_b64 %0, %1, %2"
               :: "v"(lds_off), "v"(gl_off),
                  "s"((unsigned long long)(size_t)base) : "memory");
}
DEVINL void wait_async0() { asm volatile("s_wait_asynccnt 0x0" ::: "memory"); }

// ---------------------------------------------------------------------------
// CDNA5 Tensor Data Mover: one instruction DMAs a 2D tile (rows x 64B) into
// LDS, applying the padded pitch in hardware (pad 4 DWORDs every 16 DWORDs
// -> 80B LDS pitch == LDP=40 halfs). D# built per ISA §8.3/8.4:
//   g0: [1:0]=count=1, [63:32]=lds_addr, [120:64]=global_addr, [127:126]=2
//   g1: data_size=2B, pad cfg, tensor_dim0/1 (OOB), tile_dim0/1, dim0 stride
// This toolchain exposes the 6-arg builtin:
//   (uint32x4 g0, int32x8 g1, int32x4 g2, int32x4 g3, int32x8, i32 cpol)
// ---------------------------------------------------------------------------
#if __has_builtin(__builtin_amdgcn_tensor_load_to_lds)
#define HAVE_TDM 1
typedef __attribute__((ext_vector_type(4))) unsigned int v4u;
typedef __attribute__((ext_vector_type(4))) int          v4i;
typedef __attribute__((ext_vector_type(8))) int          v8i;

DEVINL void tdm_load_2d(unsigned lds_addr, const void* gptr,
                        unsigned tile_k, unsigned tile_m,
                        unsigned tens_k, unsigned tens_m, unsigned stride_k) {
  unsigned long long ga = (unsigned long long)(size_t)gptr;
  v4u g0;
  g0[0] = 1u;                                                  // count=1
  g0[1] = lds_addr;                                            // lds_addr
  g0[2] = (unsigned)ga;                                        // gaddr[31:0]
  g0[3] = (unsigned)((ga >> 32) & 0x01FFFFFFu) | (2u << 30);   // gaddr hi|type=2
  unsigned long long b0 = 0, b1 = 0, b2 = 0, b3 = 0;           // g1 bits 0..255
  b0 |= 1ull << 16;                                 // data_size = 2 bytes
  b0 |= 1ull << 20;                                 // pad_enable
  b0 |= 3ull << 22;                                 // pad_interval: 16 DWORDs
  b0 |= 3ull << 25;                                 // pad_amount:    4 DWORDs
  b0 |= (unsigned long long)(tens_k & 0xFFFFu) << 48;  // tensor_dim0[15:0]
  b1 |= (unsigned long long)(tens_k >> 16);            // tensor_dim0[31:16]
  b1 |= (unsigned long long)tens_m << 16;              // tensor_dim1[31:0]
  b1 |= (unsigned long long)(tile_k & 0xFFFFu) << 48;  // tile_dim0
  b2 |= (unsigned long long)(tile_m & 0xFFFFu);        // tile_dim1
  b2 |= (unsigned long long)stride_k << 32;            // dim0_stride[31:0]
  v8i g1;
  g1[0] = (int)(unsigned)b0; g1[1] = (int)(unsigned)(b0 >> 32);
  g1[2] = (int)(unsigned)b1; g1[3] = (int)(unsigned)(b1 >> 32);
  g1[4] = (int)(unsigned)b2; g1[5] = (int)(unsigned)(b2 >> 32);
  g1[6] = (int)(unsigned)b3; g1[7] = (int)(unsigned)(b3 >> 32);
  v4i z4 = {0, 0, 0, 0};                // groups 2/3: 2D tensor, dims unused
  v8i z8 = {0, 0, 0, 0, 0, 0, 0, 0};    // extra group: unused for 2D tiles
  __builtin_amdgcn_tensor_load_to_lds(g0, g1, z4, z4, z8, 0);
}
DEVINL void wait_tensor0() {
#if __has_builtin(__builtin_amdgcn_s_wait_tensorcnt)
  __builtin_amdgcn_s_wait_tensorcnt(0);
#else
  asm volatile("s_wait_tensorcnt 0x0" ::: "memory");
#endif
}
#else
#define HAVE_TDM 0
#endif

// ---------------------------------------------------------------------------
// WMMA helpers (gfx1250, wave32). Fragment layouts per CDNA5 ISA 7.12.2.
// ---------------------------------------------------------------------------
DEVINL v16h cat16(v8h lo, v8h hi) {
  return __builtin_shufflevector(lo, hi, 0,1,2,3,4,5,6,7,8,9,10,11,12,13,14,15);
}
DEVINL v8f wmma32(v16h a, v16h b, v8f c) {
  return __builtin_amdgcn_wmma_f32_16x16x32_f16(false, a, false, b, (short)0, c,
                                                false, false);
}
template<int LDP>
DEVINL v16h load_a_frag(const _Float16* base) {   // base -> [m_base][k0]
  const int lane = threadIdx.x & 31;
  const _Float16* p = base + (lane & 15) * LDP + ((lane >> 4) << 3);
  return cat16(*(const v8h*)(p), *(const v8h*)(p + 16));
}
template<int LDP>
DEVINL v16h load_b_frag(const _Float16* base) {   // base -> [n_base][k0]
  const int lane = threadIdx.x & 31;
  const _Float16* p = base + (lane & 15) * LDP + ((lane >> 4) << 3);
  return cat16(*(const v8h*)(p), *(const v8h*)(p + 16));
}
DEVINL float sigmoidf_(float x) { return 1.0f / (1.0f + __expf(-x)); }

// ---------------------------------------------------------------------------
// x preprocessing: x/255 -> f16 once (67 MB, L2 resident).
// ---------------------------------------------------------------------------
__global__ __launch_bounds__(256)
void prep_x_kernel(const float* __restrict__ x, _Float16* __restrict__ x16) {
  size_t i = ((size_t)blockIdx.x * 256 + threadIdx.x) * 4;
  float4 f = *(const float4*)(x + i);
  const float s = 1.0f / 255.0f;
  v4h o;
  o[0] = (_Float16)(f.x * s); o[1] = (_Float16)(f.y * s);
  o[2] = (_Float16)(f.z * s); o[3] = (_Float16)(f.w * s);
  *(v4h*)(x16 + i) = o;
}

// ---------------------------------------------------------------------------
// Weight prep. 1,389,568 elems = 5428 x 256.
// ---------------------------------------------------------------------------
__global__ __launch_bounds__(256)
void prep_weights_kernel(const float* __restrict__ c1w, const float* __restrict__ c2w,
                         const float* __restrict__ c3w, const float* __restrict__ fcw,
                         const float* __restrict__ wih, const float* __restrict__ whh,
                         const float* __restrict__ bih, const float* __restrict__ bhh,
                         _Float16* __restrict__ wc1, _Float16* __restrict__ wc2,
                         _Float16* __restrict__ wc3, _Float16* __restrict__ wfcp,
                         _Float16* __restrict__ wihp, _Float16* __restrict__ whhp,
                         float* __restrict__ bsum) {
  long i = (long)blockIdx.x * 256 + threadIdx.x;
  if (i < 8192)  { wc1[i] = (_Float16)c1w[i]; return; }  i -= 8192;
  if (i < 32768) { wc2[i] = (_Float16)c2w[i]; return; }  i -= 32768;
  if (i < 36864) { wc3[i] = (_Float16)c3w[i]; return; }  i -= 36864;
  if (i < 524288) {
    long o = i >> 10, k = i & 1023, hw = k >> 6, c = k & 63;
    wfcp[i] = (_Float16)fcw[o * 1024 + c * 16 + hw];
    return;
  }
  i -= 524288;
  if (i < 524288) { wihp[i] = (_Float16)wih[i]; return; } i -= 524288;
  if (i < 262144) { whhp[i] = (_Float16)whh[i]; return; } i -= 262144;
  if (i < 1024)   { bsum[i] = bih[i] + bhh[i]; }
}

// ---------------------------------------------------------------------------
// conv1 implicit GEMM: M=460800 (n,oy,ox), K=256 (ci*64+ky*8+kx), N=32.
// Irregular (im2col) rows -> per-lane ASYNC-to-LDS path, double-buffered.
// ---------------------------------------------------------------------------
__global__ __launch_bounds__(256)
void conv1_gemm_kernel(const _Float16* __restrict__ x16,
                       const _Float16* __restrict__ Wc1,
                       const float* __restrict__ bias, _Float16* __restrict__ out) {
  constexpr int BM = 256, LDP = 40, WLDP = 264, N = 32;
  __shared__ _Float16 As[2][BM * LDP];
  __shared__ _Float16 Wall[32 * WLDP];
  const int tid = threadIdx.x;
  const int wave = tid >> 5, lane = tid & 31;
  const int ln = lane & 15, hi8 = (lane >> 4) << 3;
  const int m0 = blockIdx.x * BM;

  const unsigned asBase[2] = {(unsigned)(size_t)&As[0][0], (unsigned)(size_t)&As[1][0]};
  const unsigned wBase = (unsigned)(size_t)&Wall[0];

  #pragma unroll
  for (int j = 0; j < 4; ++j) {
    int i = tid + j * 256;
    int row = i >> 5, c = (i & 31) << 3;
    async_load_b128(wBase + (unsigned)(row * WLDP + c) * 2,
                    (unsigned)(row * 256 + c) * 2, Wc1);
  }

  unsigned rowP[4], ldsL[4];
  int cj[4];
  #pragma unroll
  for (int j = 0; j < 4; ++j) {
    int i = tid + j * 256;
    int row = i >> 2;
    cj[j] = i & 3;
    int gm = m0 + row;
    int n_img = gm / 225, rem = gm % 225;
    int oy = rem / 15, ox = rem % 15;
    rowP[j] = (unsigned)(n_img * 16384 + oy * 256 + ox * 4);
    ldsL[j] = (unsigned)(row * LDP + cj[j] * 8) * 2;
  }
  auto issueA = [&](int buf, int k0) {
    #pragma unroll
    for (int j = 0; j < 4; ++j) {
      int col = k0 + cj[j] * 8;
      int ci = col >> 6, ky = (col >> 3) & 7;
      unsigned g = (rowP[j] + (unsigned)(ci * 4096 + ky * 64)) * 2;
      unsigned l = asBase[buf] + ldsL[j];
      async_load_b64(l,     g,     x16);
      async_load_b64(l + 8, g + 8, x16);
    }
  };
  issueA(0, 0);

  v8f z8 = {0.f,0.f,0.f,0.f,0.f,0.f,0.f,0.f};
  v8f acc[2][2] = {{z8, z8}, {z8, z8}};

  for (int kk = 0; kk < 8; ++kk) {
    wait_async0();
    __syncthreads();
    if (kk < 7) issueA((kk + 1) & 1, (kk + 1) * 32);
    const _Float16* as = As[kk & 1];
    const int k0 = kk * 32;
    v16h a0 = load_a_frag<LDP>(&as[(wave * 32 +  0) * LDP]);
    v16h a1 = load_a_frag<LDP>(&as[(wave * 32 + 16) * LDP]);
    v16h b0 = load_b_frag<WLDP>(&Wall[ 0 * WLDP + k0]);
    v16h b1 = load_b_frag<WLDP>(&Wall[16 * WLDP + k0]);
    acc[0][0] = wmma32(a0, b0, acc[0][0]);
    acc[0][1] = wmma32(a0, b1, acc[0][1]);
    acc[1][0] = wmma32(a1, b0, acc[1][0]);
    acc[1][1] = wmma32(a1, b1, acc[1][1]);
  }
  #pragma unroll
  for (int sm = 0; sm < 2; ++sm)
    #pragma unroll
    for (int sn = 0; sn < 2; ++sn) {
      int n = sn * 16 + ln;
      float bv = bias[n];
      #pragma unroll
      for (int r = 0; r < 8; ++r) {
        int m = m0 + wave * 32 + sm * 16 + hi8 + r;
        float v = acc[sm][sn][r] + bv;
        v = v > 0.f ? v : 0.f;
        out[(size_t)m * N + n] = (_Float16)v;
      }
    }
}

// ---------------------------------------------------------------------------
// Generic WMMA GEMM: C[M][N] = act(A[M][K] @ W[N][K]^T + bias).
// A/W tiles are regular 2D tiles -> ONE tensor_load_to_lds per tile (TDM,
// wave-0 issued, TENSORcnt), hardware-padded into the 40-half LDS pitch.
// Double-buffered: tile k+1 DMAs while tile k is computed.
// ---------------------------------------------------------------------------
template<int WM, int WN, int OUT_F32, int RELU>
__global__ __launch_bounds__(WM * WN * 32)
void gemm_wmma_kernel(const _Float16* __restrict__ A, const _Float16* __restrict__ W,
                      const float* __restrict__ bias, void* __restrict__ Cout,
                      int M, int N, int K) {
  constexpr int BM = WM * 32, BN = WN * 32, NT = WM * WN * 32, LDP = 40;
  static_assert((BM * 4) % NT == 0 && (BN * 4) == NT, "tile/thread mismatch");
  __shared__ _Float16 As[2][BM * LDP];
  __shared__ _Float16 Ws[2][BN * LDP];
  const int tid = threadIdx.x;
  const int wave = tid >> 5, lane = tid & 31;
  const int wm = wave % WM, wn = wave / WM;
  const int ln = lane & 15, hi8 = (lane >> 4) << 3;
  const int m0 = blockIdx.x * BM;
  const int n0 = blockIdx.y * BN;

  const unsigned asBase[2] = {(unsigned)(size_t)&As[0][0], (unsigned)(size_t)&As[1][0]};
  const unsigned wsBase[2] = {(unsigned)(size_t)&Ws[0][0], (unsigned)(size_t)&Ws[1][0]};

#if HAVE_TDM
  auto issue_tile = [&](int buf, int k0) {
    if (wave == 0) {
      tdm_load_2d(asBase[buf], A + (size_t)m0 * K + k0, 32u, (unsigned)BM,
                  (unsigned)K, (unsigned)M, (unsigned)K);
      tdm_load_2d(wsBase[buf], W + (size_t)n0 * K + k0, 32u, (unsigned)BN,
                  (unsigned)K, (unsigned)N, (unsigned)K);
    }
  };
#else
  constexpr int ACH = (BM * 4) / NT;
  unsigned aG[ACH], aL[ACH];
  #pragma unroll
  for (int j = 0; j < ACH; ++j) {
    int i = tid + j * NT;
    int row = i >> 2, c = (i & 3) << 3;
    aG[j] = (unsigned)(((size_t)(m0 + row) * K + c) * 2);
    aL[j] = (unsigned)(row * LDP + c) * 2;
  }
  const int wrow = tid >> 2, wc = (tid & 3) << 3;
  const unsigned wG = (unsigned)(((size_t)(n0 + wrow) * K + wc) * 2);
  const unsigned wL = (unsigned)(wrow * LDP + wc) * 2;
  auto issue_tile = [&](int buf, int k0) {
    unsigned kb = (unsigned)(k0 * 2);
    #pragma unroll
    for (int j = 0; j < ACH; ++j)
      async_load_b128(asBase[buf] + aL[j], aG[j] + kb, A);
    async_load_b128(wsBase[buf] + wL, wG + kb, W);
  };
#endif
  issue_tile(0, 0);

  v8f z8 = {0.f,0.f,0.f,0.f,0.f,0.f,0.f,0.f};
  v8f acc[2][2] = {{z8, z8}, {z8, z8}};

  const int nk = K >> 5;
  for (int kk = 0; kk < nk; ++kk) {
#if HAVE_TDM
    if (wave == 0) wait_tensor0();
#else
    wait_async0();
#endif
    __syncthreads();
    if (kk + 1 < nk) issue_tile((kk + 1) & 1, (kk + 1) * 32);
    const _Float16* as = As[kk & 1];
    const _Float16* ws = Ws[kk & 1];
    v16h a0 = load_a_frag<LDP>(&as[(wm * 32 +  0) * LDP]);
    v16h a1 = load_a_frag<LDP>(&as[(wm * 32 + 16) * LDP]);
    v16h b0 = load_b_frag<LDP>(&ws[(wn * 32 +  0) * LDP]);
    v16h b1 = load_b_frag<LDP>(&ws[(wn * 32 + 16) * LDP]);
    acc[0][0] = wmma32(a0, b0, acc[0][0]);
    acc[0][1] = wmma32(a0, b1, acc[0][1]);
    acc[1][0] = wmma32(a1, b0, acc[1][0]);
    acc[1][1] = wmma32(a1, b1, acc[1][1]);
  }
  #pragma unroll
  for (int sm = 0; sm < 2; ++sm)
    #pragma unroll
    for (int sn = 0; sn < 2; ++sn) {
      int n = n0 + wn * 32 + sn * 16 + ln;
      float bv = bias ? bias[n] : 0.0f;
      #pragma unroll
      for (int r = 0; r < 8; ++r) {
        int m = m0 + wm * 32 + sm * 16 + hi8 + r;
        float v = acc[sm][sn][r] + bv;
        if (RELU) v = v > 0.f ? v : 0.f;
        if (OUT_F32) ((float*)Cout)[(size_t)m * N + n] = v;
        else         ((_Float16*)Cout)[(size_t)m * N + n] = (_Float16)v;
      }
    }
}

// ---------------------------------------------------------------------------
// im2col for conv2 (NHWC 2048x15x15x32 -> [73728][512]) and conv3.
// ---------------------------------------------------------------------------
__global__ __launch_bounds__(256)
void im2col2_kernel(const _Float16* __restrict__ act1, _Float16* __restrict__ A2) {
  size_t idx = (size_t)blockIdx.x * 256 + threadIdx.x;
  if (idx >= (size_t)73728 * 512) return;
  int col = idx & 511;
  size_t row = idx >> 9;
  int n = (int)(row / 36), rem = (int)(row % 36);
  int oy = rem / 6, ox = rem % 6;
  int ci = col >> 4, ky = (col >> 2) & 3, kx = col & 3;
  int iy = oy * 2 + ky, ix = ox * 2 + kx;
  A2[idx] = act1[(((size_t)n * 15 + iy) * 15 + ix) * 32 + ci];
}

__global__ __launch_bounds__(256)
void im2col3_kernel(const _Float16* __restrict__ act2, _Float16* __restrict__ A3) {
  size_t idx = (size_t)blockIdx.x * 256 + threadIdx.x;
  if (idx >= (size_t)32768 * 576) return;
  int col = (int)(idx % 576);
  size_t row = idx / 576;
  int n = (int)(row >> 4), rem = (int)(row & 15);
  int oy = rem >> 2, ox = rem & 3;
  int ci = col / 9, r9 = col % 9, ky = r9 / 3, kx = r9 % 3;
  int iy = oy + ky, ix = ox + kx;
  A3[idx] = act2[(((size_t)n * 6 + iy) * 6 + ix) * 64 + ci];
}

// ---------------------------------------------------------------------------
// LSTM recurrence: one WG, 512 threads = 16 waves; wave w owns units
// [w*16,w*16+16) and computes ALL 4 gates for them -> i,f,g,o for the same
// (row,unit) land in one lane; c-state lives in VGPRs across 64 steps.
// ---------------------------------------------------------------------------
__global__ __launch_bounds__(512)
void lstm_kernel(const float* __restrict__ gx, const _Float16* __restrict__ whh,
                 const int* __restrict__ done, const float* __restrict__ h0,
                 const float* __restrict__ c0, float* __restrict__ out) {
  constexpr int LDP = 264;
  __shared__ _Float16 h16[32 * LDP];
  const int tid = threadIdx.x, lane = tid & 31, wave = tid >> 5;
  const int ln = lane & 15, hi8 = (lane >> 4) << 3;
  const int u = wave * 16 + ln;

  float h_reg[2][8], c_reg[2][8];
  #pragma unroll
  for (int sm = 0; sm < 2; ++sm)
    #pragma unroll
    for (int r = 0; r < 8; ++r) {
      int m = sm * 16 + hi8 + r;
      h_reg[sm][r] = h0[(size_t)m * 256 + u];
      c_reg[sm][r] = c0[(size_t)m * 256 + u];
    }

  for (int t = 0; t < 64; ++t) {
    #pragma unroll
    for (int sm = 0; sm < 2; ++sm)
      #pragma unroll
      for (int r = 0; r < 8; ++r) {
        int m = sm * 16 + hi8 + r;
        float mk = 1.0f - (float)done[t * 32 + m];
        c_reg[sm][r] *= mk;
        h16[m * LDP + u] = (_Float16)(h_reg[sm][r] * mk);
      }
    __syncthreads();

    v8f acc[2][4];
    #pragma unroll
    for (int sm = 0; sm < 2; ++sm)
      #pragma unroll
      for (int g = 0; g < 4; ++g) {
        int nn = g * 256 + wave * 16 + ln;
        #pragma unroll
        for (int r = 0; r < 8; ++r)
          acc[sm][g][r] = gx[(size_t)(t * 32 + sm * 16 + hi8 + r) * 1024 + nn];
      }
    for (int k0 = 0; k0 < 256; k0 += 32) {
      v16h a0 = load_a_frag<LDP>(&h16[ 0 * LDP + k0]);
      v16h a1 = load_a_frag<LDP>(&h16[16 * LDP + k0]);
      #pragma unroll
      for (int g = 0; g < 4; ++g) {
        v16h b = load_b_frag<256>(whh + (size_t)(g * 256 + wave * 16) * 256 + k0);
        acc[0][g] = wmma32(a0, b, acc[0][g]);
        acc[1][g] = wmma32(a1, b, acc[1][g]);
      }
    }
    __syncthreads();

    #pragma unroll
    for (int sm = 0; sm < 2; ++sm)
      #pragma unroll
      for (int r = 0; r < 8; ++r) {
        float ig = sigmoidf_(acc[sm][0][r]);
        float fg = sigmoidf_(acc[sm][1][r]);
        float gg = tanhf(acc[sm][2][r]);
        float og = sigmoidf_(acc[sm][3][r]);
        float cn = fg * c_reg[sm][r] + ig * gg;
        float hn = og * tanhf(cn);
        c_reg[sm][r] = cn;
        h_reg[sm][r] = hn;
        int m = sm * 16 + hi8 + r;
        out[(size_t)(t * 32 + m) * 256 + u] = hn;
      }
  }
}

// ---------------------------------------------------------------------------
extern "C" void kernel_launch(void* const* d_in, const int* in_sizes, int n_in,
                              void* d_out, int out_size, void* d_ws, size_t ws_size,
                              hipStream_t stream) {
  const float* x   = (const float*)d_in[0];
  const int*   dn  = (const int*)  d_in[1];
  const float* c1w = (const float*)d_in[2];
  const float* c1b = (const float*)d_in[3];
  const float* c2w = (const float*)d_in[4];
  const float* c2b = (const float*)d_in[5];
  const float* c3w = (const float*)d_in[6];
  const float* c3b = (const float*)d_in[7];
  const float* fcw = (const float*)d_in[8];
  const float* fcb = (const float*)d_in[9];
  const float* wih = (const float*)d_in[10];
  const float* whh = (const float*)d_in[11];
  const float* bih = (const float*)d_in[12];
  const float* bhh = (const float*)d_in[13];
  const float* h0  = (const float*)d_in[14];
  const float* c0  = (const float*)d_in[15];
  float* out = (float*)d_out;

  char* ws = (char*)d_ws;
  size_t off = 0;
  auto alloc = [&](size_t bytes) -> char* {
    char* p = ws + off;
    off = (off + bytes + 255) & ~(size_t)255;
    return p;
  };
  _Float16* wc1  = (_Float16*)alloc((size_t)8192 * 2);
  _Float16* wc2  = (_Float16*)alloc((size_t)32768 * 2);
  _Float16* wc3  = (_Float16*)alloc((size_t)36864 * 2);
  _Float16* wfcp = (_Float16*)alloc((size_t)524288 * 2);
  _Float16* wihp = (_Float16*)alloc((size_t)524288 * 2);
  _Float16* whhp = (_Float16*)alloc((size_t)262144 * 2);
  float*    bsum = (float*)   alloc((size_t)1024 * 4);
  _Float16* x16  = (_Float16*)alloc((size_t)33554432 * 2);
  _Float16* col  = (_Float16*)alloc((size_t)73728 * 512 * 2);
  _Float16* act1 = (_Float16*)alloc((size_t)460800 * 32 * 2);
  _Float16* act2 = (_Float16*)alloc((size_t)73728 * 64 * 2);
  _Float16* act3 = (_Float16*)alloc((size_t)32768 * 64 * 2);
  _Float16* fco  = (_Float16*)alloc((size_t)2048 * 512 * 2);
  float*    gx   = (float*)   alloc((size_t)2048 * 1024 * 4);

  prep_x_kernel<<<32768, 256, 0, stream>>>(x, x16);
  prep_weights_kernel<<<5428, 256, 0, stream>>>(c1w, c2w, c3w, fcw, wih, whh,
                                                bih, bhh, wc1, wc2, wc3, wfcp,
                                                wihp, whhp, bsum);
  conv1_gemm_kernel<<<1800, 256, 0, stream>>>(x16, wc1, c1b, act1);
  im2col2_kernel<<<147456, 256, 0, stream>>>(act1, col);
  gemm_wmma_kernel<4, 2, 0, 1><<<dim3(576, 1), 256, 0, stream>>>(
      col, wc2, c2b, act2, 73728, 64, 512);
  im2col3_kernel<<<73728, 256, 0, stream>>>(act2, col);
  gemm_wmma_kernel<4, 2, 0, 1><<<dim3(256, 1), 256, 0, stream>>>(
      col, wc3, c3b, act3, 32768, 64, 576);
  gemm_wmma_kernel<4, 2, 0, 1><<<dim3(16, 8), 256, 0, stream>>>(
      act3, wfcp, fcb, fco, 2048, 512, 1024);
  gemm_wmma_kernel<4, 2, 1, 0><<<dim3(16, 16), 256, 0, stream>>>(
      fco, wihp, bsum, gx, 2048, 1024, 512);
  lstm_kernel<<<1, 512, 0, stream>>>(gx, whhp, dn, h0, c0, out);

  (void)in_sizes; (void)n_in; (void)out_size; (void)ws_size;
}
